// HolleyGAT_58686433133219
// MI455X (gfx1250) — matrix operand order repaired and checked by reading, not compiled, not run on hardware
//
#include <hip/hip_runtime.h>
#include <hip/hip_bf16.h>

typedef __bf16 bf16;
typedef __bf16 bf16x8 __attribute__((ext_vector_type(8)));
typedef __bf16 v16bf  __attribute__((ext_vector_type(16)));
typedef float  v8f    __attribute__((ext_vector_type(8)));

#define N_USER 100000
#define N_PROD 50000
#define N_VEH  10000
#define HID    256
#define NEG_SLOPE 0.2f

// ---------------- device helpers ----------------

__device__ __forceinline__ unsigned f2ord(float f) {
    unsigned u = __float_as_uint(f);
    return (u & 0x80000000u) ? ~u : (u | 0x80000000u);
}
__device__ __forceinline__ float ord2f(unsigned u) {
    unsigned v = (u & 0x80000000u) ? (u ^ 0x80000000u) : ~u;
    return __uint_as_float(v);
}
__device__ __forceinline__ float leaky(float x) {
    return x > 0.f ? x : NEG_SLOPE * x;
}

// CDNA5 async global->LDS copy (per-lane 16B), tracked by ASYNCcnt.
__device__ __forceinline__ void async_load_b128(unsigned lds_byte_off, const void* gptr) {
    asm volatile("global_load_async_to_lds_b128 %0, %1, off"
                 :: "v"(lds_byte_off), "v"((unsigned long long)(size_t)gptr)
                 : "memory");
}
__device__ __forceinline__ void wait_asynccnt0() {
    asm volatile("s_wait_asynccnt 0" ::: "memory");
}

// ---------------- elementwise kernels ----------------

__global__ void fill_u32_kernel(unsigned* p, unsigned v, size_t n) {
    size_t i = (size_t)blockIdx.x * blockDim.x + threadIdx.x;
    if (i < n) p[i] = v;
}

__global__ void f32_to_bf16_kernel(const float* __restrict__ x, bf16* __restrict__ y, size_t n) {
    size_t i = (size_t)blockIdx.x * blockDim.x + threadIdx.x;
    if (i < n) y[i] = (bf16)x[i];
}

__global__ void elu_to_bf16_kernel(const float* __restrict__ x, bf16* __restrict__ y, size_t n) {
    size_t i = (size_t)blockIdx.x * blockDim.x + threadIdx.x;
    if (i < n) {
        float v = x[i];
        v = v > 0.f ? v : (__expf(v) - 1.f);
        y[i] = (bf16)v;
    }
}

__global__ void add_bias_kernel(float* __restrict__ acc, const float* __restrict__ b, size_t n) {
    size_t i = (size_t)blockIdx.x * blockDim.x + threadIdx.x;
    if (i < n) acc[i] += b[i & (HID - 1)];
}

// ---------------- product-feature MLP ----------------
__global__ void mlp_product_kernel(const int* __restrict__ part_type,
                                   const float* __restrict__ prod_feat,
                                   const float* __restrict__ pt_emb,
                                   const float* __restrict__ w1, const float* __restrict__ b1,
                                   const float* __restrict__ w2, const float* __restrict__ b2,
                                   const float* __restrict__ prod_emb,
                                   float* __restrict__ out) {
    __shared__ float feat[35];
    __shared__ float hbuf[128];
    int n = blockIdx.x;
    int t = threadIdx.x;
    if (t < 32)       feat[t] = pt_emb[(size_t)part_type[n] * 32 + t];
    else if (t < 35)  feat[t] = prod_feat[(size_t)n * 3 + (t - 32)];
    __syncthreads();
    float s = b1[t];
    #pragma unroll 5
    for (int k = 0; k < 35; ++k) s += feat[k] * w1[k * 128 + t];
    s = fmaxf(s, 0.f);
    hbuf[t] = s;
    __syncthreads();
    float o = b2[t];
    for (int k = 0; k < 128; ++k) o += hbuf[k] * w2[k * 128 + t];
    out[(size_t)n * 128 + t] = prod_emb[(size_t)n * 128 + t] + o;
}

// ---------------- weight prep ----------------
// Wt[n][k] = (bf16) W[k][n]   (W is [K, NOUT] row-major)
__global__ void wt_convert_kernel(const float* __restrict__ W, bf16* __restrict__ Wt,
                                  int K, int NOUT) {
    int i = blockIdx.x * blockDim.x + threadIdx.x;
    if (i >= K * NOUT) return;
    int n = i / K, k = i - n * K;
    Wt[i] = (bf16)W[(size_t)k * NOUT + n];
}

// watt_{s,d}[k][h] = sum_c W[k][h*64+c] * a_{src,dst}[h][c]
__global__ void build_watt_kernel(const float* __restrict__ W,
                                  const float* __restrict__ a_src, const float* __restrict__ a_dst,
                                  float* __restrict__ watt_s, float* __restrict__ watt_d, int K) {
    int i = blockIdx.x * blockDim.x + threadIdx.x;
    if (i >= K * 4) return;
    int k = i >> 2, h = i & 3;
    const float* wr = W + (size_t)k * HID + h * 64;
    float ss = 0.f, sd = 0.f;
    for (int c = 0; c < 64; ++c) { ss += wr[c] * a_src[h * 64 + c]; sd += wr[c] * a_dst[h * 64 + c]; }
    watt_s[k * 4 + h] = ss;
    watt_d[k * 4 + h] = sd;
}

// ---------------- WMMA GEMM: Y[N,NOUT] = Xb[N,K] @ Wt[NOUT,K]^T  (bf16 in, f32 acc/out) ---
// Block: 256 threads = 8 waves; block tile = 128(M) x 64(N).
// B slab (64 cols x K) staged once per block into LDS via async global->LDS copies;
// A fragments double-buffered in registers. K % 32 == 0, NOUT % 64 == 0, K <= 256.
__global__ void __launch_bounds__(256)
gemm_bf16_wmma_kernel(const bf16* __restrict__ X, const bf16* __restrict__ Wt,
                      float* __restrict__ Y, const float* __restrict__ bias,
                      int Nrows, int K, int NOUT, int act) {
    __shared__ bf16 lds_wt[64 * 256];                    // up to 32 KB
    const int tid  = threadIdx.x;
    const int wave = tid >> 5;
    const int lane = tid & 31;
    const int colBase = blockIdx.y * 64;

    // ---- async stage of B slab: rows [colBase, colBase+64) of Wt, all K ----
    {
        const bf16* slab = Wt + (size_t)colBase * K;      // 64*K contiguous bf16
        unsigned lds_base = (unsigned)(size_t)(void*)lds_wt;
        const int nchunk = (64 * K) / 8;                  // 16-byte chunks
        for (int ch = tid; ch < nchunk; ch += 256)
            async_load_b128(lds_base + (unsigned)ch * 16u, slab + (size_t)ch * 8);
        wait_asynccnt0();
        __syncthreads();
    }

    const int tileM = blockIdx.x * 8 + wave;
    const int g  = lane >> 4;                             // lane group (0/1)
    const int rl = lane & 15;
    int row  = tileM * 16 + rl;
    int rowc = row < Nrows ? row : (Nrows - 1);           // clamp loads, guard stores
    const bf16* xp = X + (size_t)rowc * K;

    v8f acc[4] = {};
    v16bf a_cur;                                          // A: row rl, K chunks 8g and 16+8g
    ((bf16x8*)&a_cur)[0] = *(const bf16x8*)(xp + 8 * g);
    ((bf16x8*)&a_cur)[1] = *(const bf16x8*)(xp + 16 + 8 * g);

    for (int k0 = 0; k0 < K; k0 += 32) {
        v16bf a_nxt;
        if (k0 + 32 < K) {                                // prefetch next A fragment
            ((bf16x8*)&a_nxt)[0] = *(const bf16x8*)(xp + k0 + 32 + 8 * g);
            ((bf16x8*)&a_nxt)[1] = *(const bf16x8*)(xp + k0 + 48 + 8 * g);
        }
        #pragma unroll
        for (int nt = 0; nt < 4; ++nt) {
            const bf16* wp = &lds_wt[(nt * 16 + rl) * K + k0 + 8 * g];
            v16bf bfrag;                                  // B: col rl, K chunks 8g and 16+8g
            ((bf16x8*)&bfrag)[0] = *(const bf16x8*)(wp);
            ((bf16x8*)&bfrag)[1] = *(const bf16x8*)(wp + 16);
            acc[nt] = __builtin_amdgcn_wmma_f32_16x16x32_bf16(
                false, a_cur, false, bfrag, (short)0, acc[nt], false, false);
        }
        if (k0 + 32 < K) a_cur = a_nxt;
    }

    #pragma unroll
    for (int nt = 0; nt < 4; ++nt) {
        int col = colBase + nt * 16 + rl;
        float bv = bias ? bias[col] : 0.f;
        #pragma unroll
        for (int j = 0; j < 8; ++j) {
            int r = tileM * 16 + g * 8 + j;               // D: M = j + 8*lanegroup
            if (r < Nrows) {
                float v = acc[nt][j] + bv;
                if (act) v = fmaxf(v, 0.f);
                Y[(size_t)r * NOUT + col] = v;
            }
        }
    }
}

// ---------------- attention logits: A[n][h] = sum_k X[n][k] * watt[k][h] ----------------
__global__ void logits_kernel(const bf16* __restrict__ X, const float* __restrict__ watt,
                              float* __restrict__ A, int N, int K) {
    int lane = threadIdx.x & 31;
    int n = blockIdx.x * 8 + (threadIdx.x >> 5);
    if (n >= N) return;
    const bf16* xp = X + (size_t)n * K;
    float s0 = 0.f, s1 = 0.f, s2 = 0.f, s3 = 0.f;
    for (int k = lane; k < K; k += 32) {
        float xv = (float)xp[k];
        float4 w = ((const float4*)watt)[k];
        s0 += xv * w.x; s1 += xv * w.y; s2 += xv * w.z; s3 += xv * w.w;
    }
    for (int off = 16; off; off >>= 1) {
        s0 += __shfl_xor(s0, off, 32);
        s1 += __shfl_xor(s1, off, 32);
        s2 += __shfl_xor(s2, off, 32);
        s3 += __shfl_xor(s3, off, 32);
    }
    if (lane == 0) { float4 r = {s0, s1, s2, s3}; ((float4*)A)[n] = r; }
}

// ---------------- edge passes ----------------

__global__ void edge_max_kernel(const int* __restrict__ src, const int* __restrict__ dst,
                                const float* __restrict__ As, const float* __restrict__ Ad,
                                unsigned* __restrict__ M, int E) {
    int e = blockIdx.x * blockDim.x + threadIdx.x;
    if (e >= E) return;
    int s = src[e], d = dst[e];
    float4 as = ((const float4*)As)[s];
    float4 ad = ((const float4*)Ad)[d];
    atomicMax(&M[d * 4 + 0], f2ord(leaky(as.x + ad.x)));
    atomicMax(&M[d * 4 + 1], f2ord(leaky(as.y + ad.y)));
    atomicMax(&M[d * 4 + 2], f2ord(leaky(as.z + ad.z)));
    atomicMax(&M[d * 4 + 3], f2ord(leaky(as.w + ad.w)));
}

__global__ void edge_den_kernel(const int* __restrict__ src, const int* __restrict__ dst,
                                const float* __restrict__ As, const float* __restrict__ Ad,
                                const unsigned* __restrict__ M, float* __restrict__ Den, int E) {
    int e = blockIdx.x * blockDim.x + threadIdx.x;
    if (e >= E) return;
    int s = src[e], d = dst[e];
    float4 as = ((const float4*)As)[s];
    float4 ad = ((const float4*)Ad)[d];
    atomicAdd(&Den[d * 4 + 0], __expf(leaky(as.x + ad.x) - ord2f(M[d * 4 + 0])));
    atomicAdd(&Den[d * 4 + 1], __expf(leaky(as.y + ad.y) - ord2f(M[d * 4 + 1])));
    atomicAdd(&Den[d * 4 + 2], __expf(leaky(as.z + ad.z) - ord2f(M[d * 4 + 2])));
    atomicAdd(&Den[d * 4 + 3], __expf(leaky(as.w + ad.w) - ord2f(M[d * 4 + 3])));
}

// one wave per edge: scatter (exp(e-m)/den) * h_src[src] into accumulator (256 floats)
__global__ void edge_scatter_kernel(const int* __restrict__ src, const int* __restrict__ dst,
                                    const float* __restrict__ As, const float* __restrict__ Ad,
                                    const unsigned* __restrict__ M, const float* __restrict__ Den,
                                    const float* __restrict__ Hsrc, float* __restrict__ Acc, int E) {
    int lane = threadIdx.x & 31;
    int e = blockIdx.x * 8 + (threadIdx.x >> 5);
    if (e >= E) return;
    int s = src[e], d = dst[e];
    int h = lane >> 3;                                   // 8 lanes (64 cols) per head
    float a  = leaky(As[s * 4 + h] + Ad[d * 4 + h]);
    float m  = ord2f(M[d * 4 + h]);
    float dn = fmaxf(Den[d * 4 + h], 1e-16f);
    float w  = __expf(a - m) / dn;
    const float4* hp = (const float4*)(Hsrc + (size_t)s * HID + lane * 8);
    float4 h0 = hp[0], h1 = hp[1];
    float* ap = Acc + (size_t)d * HID + lane * 8;
    atomicAdd(ap + 0, w * h0.x); atomicAdd(ap + 1, w * h0.y);
    atomicAdd(ap + 2, w * h0.z); atomicAdd(ap + 3, w * h0.w);
    atomicAdd(ap + 4, w * h1.x); atomicAdd(ap + 5, w * h1.y);
    atomicAdd(ap + 6, w * h1.z); atomicAdd(ap + 7, w * h1.w);
}

// ---------------- L2 normalize rows of 128 ----------------
__global__ void l2norm_kernel(const float* __restrict__ X, float* __restrict__ out, int N) {
    int lane = threadIdx.x & 31;
    int n = blockIdx.x * 8 + (threadIdx.x >> 5);
    if (n >= N) return;
    float4 v = ((const float4*)(X + (size_t)n * 128))[lane];
    float s = v.x * v.x + v.y * v.y + v.z * v.z + v.w * v.w;
    for (int off = 16; off; off >>= 1) s += __shfl_xor(s, off, 32);
    float inv = 1.f / fmaxf(sqrtf(s), 1e-12f);
    float4 r = {v.x * inv, v.y * inv, v.z * inv, v.w * inv};
    ((float4*)(out + (size_t)n * 128))[lane] = r;
}

// ---------------- host orchestration ----------------

static inline unsigned ceil_div(size_t a, unsigned b) { return (unsigned)((a + b - 1) / b); }

extern "C" void kernel_launch(void* const* d_in, const int* in_sizes, int n_in,
                              void* d_out, int out_size, void* d_ws, size_t ws_size,
                              hipStream_t stream) {
    (void)n_in; (void)out_size; (void)ws_size;
    const int* e_int = (const int*)d_in[0];
    const int* e_fit = (const int*)d_in[1];
    const int* e_own = (const int*)d_in[2];
    const int* e_co  = (const int*)d_in[3];
    const int* part_type = (const int*)d_in[4];
    const float* prod_feat = (const float*)d_in[5];
    const int Eint = in_sizes[0] / 2, Efit = in_sizes[1] / 2;
    const int Eown = in_sizes[2] / 2, Eco  = in_sizes[3] / 2;
    #define PF(i) ((const float*)d_in[i])
    const float* user_emb = PF(6);  const float* product_emb = PF(7);
    const float* veh_emb  = PF(8);  const float* pt_emb = PF(9);
    const float* mlp_w1 = PF(10); const float* mlp_b1 = PF(11);
    const float* mlp_w2 = PF(12); const float* mlp_b2 = PF(13);
    // conv1 leaves: 14..41, conv2: 42..69, user_tower: 70..73, product_tower: 74..77

    // workspace carve-out
    char* wsb = (char*)d_ws;
    size_t off = 0;
    auto take = [&](size_t bytes) -> void* {
        void* p = wsb + off;
        off = (off + bytes + 255) & ~(size_t)255;
        return p;
    };
    float* prod_x   = (float*)take((size_t)N_PROD * 128 * 4);
    bf16*  xb[3];
    xb[0] = (bf16*)take((size_t)N_USER * HID * 2);
    xb[1] = (bf16*)take((size_t)N_PROD * HID * 2);
    xb[2] = (bf16*)take((size_t)N_VEH  * HID * 2);
    float* acc[3];
    acc[0] = (float*)take((size_t)N_USER * HID * 4);
    acc[1] = (float*)take((size_t)N_PROD * HID * 4);
    acc[2] = (float*)take((size_t)N_VEH  * HID * 4);
    float*    hsrc  = (float*)take((size_t)N_USER * HID * 4);   // per-relation h_src / tower scratch
    float*    a_s   = (float*)take((size_t)N_USER * 4 * 4);
    float*    a_d   = (float*)take((size_t)N_USER * 4 * 4);
    unsigned* mbuf  = (unsigned*)take((size_t)N_USER * 4 * 4);
    float*    den   = (float*)take((size_t)N_USER * 4 * 4);
    bf16*     wt    = (bf16*)take((size_t)HID * HID * 2);
    float*    watt_s = (float*)take((size_t)HID * 4 * 4);
    float*    watt_d = (float*)take((size_t)HID * 4 * 4);

    const int nnodes[3] = {N_USER, N_PROD, N_VEH};

    struct RelDesc { const int* src; const int* dst; int E; int stype; int dtype; };
    RelDesc rels[7] = {
        { e_int,        e_int + Eint, Eint, 0, 1 },  // interacts      user -> product
        { e_int + Eint, e_int,        Eint, 1, 0 },  // rev_interacts
        { e_fit,        e_fit + Efit, Efit, 1, 2 },  // fits           product -> vehicle
        { e_fit + Efit, e_fit,        Efit, 2, 1 },  // rev_fits
        { e_own,        e_own + Eown, Eown, 0, 2 },  // owns           user -> vehicle
        { e_own + Eown, e_own,        Eown, 2, 0 },  // rev_owns
        { e_co,         e_co + Eco,   Eco,  1, 1 },  // co_purchased   product -> product
    };

    // Stage 0: product MLP + bf16 conversions of layer-1 inputs (din = 128)
    mlp_product_kernel<<<N_PROD, 128, 0, stream>>>(part_type, prod_feat, pt_emb,
                                                   mlp_w1, mlp_b1, mlp_w2, mlp_b2,
                                                   product_emb, prod_x);
    f32_to_bf16_kernel<<<ceil_div((size_t)N_USER * 128, 256), 256, 0, stream>>>(user_emb, xb[0], (size_t)N_USER * 128);
    f32_to_bf16_kernel<<<ceil_div((size_t)N_PROD * 128, 256), 256, 0, stream>>>(prod_x,   xb[1], (size_t)N_PROD * 128);
    f32_to_bf16_kernel<<<ceil_div((size_t)N_VEH  * 128, 256), 256, 0, stream>>>(veh_emb,  xb[2], (size_t)N_VEH  * 128);

    // Two hetero-GAT layers
    for (int layer = 0; layer < 2; ++layer) {
        const int K = layer ? HID : 128;
        const int pbase = layer ? 42 : 14;
        for (int t = 0; t < 3; ++t)
            fill_u32_kernel<<<ceil_div((size_t)nnodes[t] * HID, 256), 256, 0, stream>>>(
                (unsigned*)acc[t], 0u, (size_t)nnodes[t] * HID);

        for (int r = 0; r < 7; ++r) {
            const RelDesc& R = rels[r];
            const float* W  = PF(pbase + r * 4 + 0);
            const float* as = PF(pbase + r * 4 + 1);
            const float* ad = PF(pbase + r * 4 + 2);
            const float* b  = PF(pbase + r * 4 + 3);
            const int nsrc = nnodes[R.stype], ndst = nnodes[R.dtype];

            wt_convert_kernel<<<ceil_div((size_t)K * HID, 256), 256, 0, stream>>>(W, wt, K, HID);
            build_watt_kernel<<<ceil_div((size_t)K * 4, 256), 256, 0, stream>>>(W, as, ad, watt_s, watt_d, K);

            dim3 gg(ceil_div((size_t)(nsrc + 15) / 16, 8), HID / 64);
            gemm_bf16_wmma_kernel<<<gg, 256, 0, stream>>>(xb[R.stype], wt, hsrc, nullptr, nsrc, K, HID, 0);

            logits_kernel<<<ceil_div((size_t)nsrc, 8), 256, 0, stream>>>(xb[R.stype], watt_s, a_s, nsrc, K);
            logits_kernel<<<ceil_div((size_t)ndst, 8), 256, 0, stream>>>(xb[R.dtype], watt_d, a_d, ndst, K);

            fill_u32_kernel<<<ceil_div((size_t)ndst * 4, 256), 256, 0, stream>>>(mbuf, 0x007FFFFFu /* ord(-inf) */, (size_t)ndst * 4);
            fill_u32_kernel<<<ceil_div((size_t)ndst * 4, 256), 256, 0, stream>>>((unsigned*)den, 0u, (size_t)ndst * 4);

            edge_max_kernel<<<ceil_div((size_t)R.E, 256), 256, 0, stream>>>(R.src, R.dst, a_s, a_d, mbuf, R.E);
            edge_den_kernel<<<ceil_div((size_t)R.E, 256), 256, 0, stream>>>(R.src, R.dst, a_s, a_d, mbuf, den, R.E);
            edge_scatter_kernel<<<ceil_div((size_t)R.E, 8), 256, 0, stream>>>(R.src, R.dst, a_s, a_d, mbuf, den,
                                                                              hsrc, acc[R.dtype], R.E);
            add_bias_kernel<<<ceil_div((size_t)ndst * HID, 256), 256, 0, stream>>>(acc[R.dtype], b, (size_t)ndst * HID);
        }
        // ELU + bf16 conversion (input of next stage, din = 256)
        for (int t = 0; t < 3; ++t)
            elu_to_bf16_kernel<<<ceil_div((size_t)nnodes[t] * HID, 256), 256, 0, stream>>>(
                acc[t], xb[t], (size_t)nnodes[t] * HID);
    }

    // Towers + L2 normalize -> d_out (u then p)
    float* out = (float*)d_out;
    const int tower_base[2] = {70, 74};
    const int tower_n[2]    = {N_USER, N_PROD};
    size_t out_off = 0;
    for (int tw = 0; tw < 2; ++tw) {
        const float* w1 = PF(tower_base[tw] + 0);
        const float* b1 = PF(tower_base[tw] + 1);
        const float* w2 = PF(tower_base[tw] + 2);
        const float* b2 = PF(tower_base[tw] + 3);
        const int N = tower_n[tw];
        bf16* hb = (bf16*)acc[tw];   // accumulators are dead now -> bf16 hidden scratch

        wt_convert_kernel<<<ceil_div((size_t)HID * HID, 256), 256, 0, stream>>>(w1, wt, HID, HID);
        dim3 g1(ceil_div((size_t)(N + 15) / 16, 8), HID / 64);
        gemm_bf16_wmma_kernel<<<g1, 256, 0, stream>>>(xb[tw], wt, hsrc, b1, N, HID, HID, 1);  // relu
        f32_to_bf16_kernel<<<ceil_div((size_t)N * HID, 256), 256, 0, stream>>>(hsrc, hb, (size_t)N * HID);

        wt_convert_kernel<<<ceil_div((size_t)HID * 128, 256), 256, 0, stream>>>(w2, wt, HID, 128);
        dim3 g2(ceil_div((size_t)(N + 15) / 16, 8), 128 / 64);
        gemm_bf16_wmma_kernel<<<g2, 256, 0, stream>>>(hb, wt, hsrc, b2, N, HID, 128, 0);

        l2norm_kernel<<<ceil_div((size_t)N, 8), 256, 0, stream>>>(hsrc, out + out_off, N);
        out_off += (size_t)N * 128;
    }
    #undef PF
}
